// GAU_75780402971292
// MI455X (gfx1250) — matrix operand order repaired and checked by reading, not compiled, not run on hardware
//
#include <hip/hip_runtime.h>

// ---------------------------------------------------------------------------
// GAU block for MI455X (gfx1250): bf16 WMMA GEMM pipeline, f32 accumulation.
// Wave tile 64x64 (16 WMMAs / 16 b128 loads per K-step), block tile 256x128.
// ---------------------------------------------------------------------------

typedef __attribute__((ext_vector_type(16))) __bf16 v16bf;
typedef __attribute__((ext_vector_type(8)))  float  v8f;

union Frag {
  v16bf v;
  uint4 q[2];
};

__device__ __forceinline__ unsigned short f2bf(float f) {
  union { float f; unsigned u; } t;
  t.f = f;
  unsigned u = t.u;
  unsigned r = u + 0x7FFFu + ((u >> 16) & 1u);   // round-to-nearest-even
  return (unsigned short)(r >> 16);
}

__device__ __forceinline__ float bf2f(unsigned short s) {
  union { float f; unsigned u; } t;
  t.u = ((unsigned)s) << 16;
  return t.f;
}

// ---------------------------------------------------------------------------
// Conversion kernels
// ---------------------------------------------------------------------------
__global__ void __launch_bounds__(256)
cvt_bf16_kernel(const float* __restrict__ in, unsigned short* __restrict__ out,
                long long n) {
  long long i = (long long)blockIdx.x * blockDim.x + threadIdx.x;
  long long stride = (long long)gridDim.x * blockDim.x;
  for (; i < n; i += stride) out[i] = f2bf(in[i]);
}

// out[c*R + r] = bf16(in[r*C + c])   (transpose during conversion)
__global__ void __launch_bounds__(256)
cvt_bf16_tr_kernel(const float* __restrict__ in, unsigned short* __restrict__ out,
                   int R, int C) {
  int i = blockIdx.x * blockDim.x + threadIdx.x;
  if (i < R * C) {
    int r = i / C, c = i % C;
    out[(size_t)c * R + r] = f2bf(in[i]);
  }
}

// ---------------------------------------------------------------------------
// Generic bf16 WMMA GEMM: C[M,N] = A[M,K] * B[N,K]^T  (B stored N x K)
// Block = 256 threads (8 wave32s) arranged 4(M) x 2(N); each wave computes a
// 64x64 tile: 4x4 accumulators, 16 v_wmma_f32_16x16x32_bf16 per K-step.
//
// EPI epilogues:
//  0: outf = acc + bias[n] (+ xres[r*N+n] if xres)             (f32 store)
//  1: h GEMM: v = relu(acc+bias); hb store; n<HID -> v^T store (bf16)
//  2: relu(acc + bias)                                         (bf16 store)
//  3: plain                                                    (bf16 store)
//  4: relu(acc*scale)^2                                        (bf16 store)
//  5: acc * gate (gate = bf16 hb[:,HID+n])                     (bf16 store)
// ---------------------------------------------------------------------------
template <int EPI>
__global__ void __launch_bounds__(256, 1)
gemm_wmma(const unsigned short* __restrict__ A,
          const unsigned short* __restrict__ Bm,
          int K,
          size_t bsA, size_t bsB, size_t bsO,
          const float* __restrict__ bias,
          const float* __restrict__ xres,
          float* __restrict__ outf,
          unsigned short* __restrict__ outb,
          unsigned short* __restrict__ vt,          // EPI 1: v^T destination
          const unsigned short* __restrict__ gateb, // EPI 5: hb base
          float scale) {
  const int N    = gridDim.x * 128;
  const int wave = threadIdx.x >> 5;
  const int lane = threadIdx.x & 31;
  const int z    = blockIdx.z;
  const int row0 = blockIdx.y * 256 + (wave >> 1) * 64;
  const int col0 = blockIdx.x * 128 + (wave & 1) * 64;
  const int r    = lane & 15;
  const int h    = lane >> 4;

  const unsigned short* Ab = A  + (size_t)z * bsA;
  const unsigned short* Bb = Bm + (size_t)z * bsB;

  // A fragment (16x32 bf16): lane-half h -> K in {8h..8h+7} U {16+8h..16+8h+7}
  const unsigned short* pa = Ab + (size_t)(row0 + r) * K + h * 8;
  // B fragment (32x16 bf16): lane-half h -> K in {8h..8h+7} U {16+8h..16+8h+7}
  const unsigned short* pb = Bb + (size_t)(col0 + r) * K + h * 8;

  v8f acc[4][4];
#pragma unroll
  for (int mi = 0; mi < 4; ++mi)
#pragma unroll
    for (int ni = 0; ni < 4; ++ni) acc[mi][ni] = (v8f){};

  for (int k = 0; k < K; k += 32) {
    Frag a[4], b[4];
#pragma unroll
    for (int mi = 0; mi < 4; ++mi) {
      const unsigned short* p = pa + (size_t)(mi * 16) * K + k;
      a[mi].q[0] = *(const uint4*)(p);
      a[mi].q[1] = *(const uint4*)(p + 16);
    }
#pragma unroll
    for (int ni = 0; ni < 4; ++ni) {
      const unsigned short* p = pb + (size_t)(ni * 16) * K + k;
      b[ni].q[0] = *(const uint4*)(p);
      b[ni].q[1] = *(const uint4*)(p + 16);
    }
#pragma unroll
    for (int mi = 0; mi < 4; ++mi)
#pragma unroll
      for (int ni = 0; ni < 4; ++ni)
        acc[mi][ni] = __builtin_amdgcn_wmma_f32_16x16x32_bf16(
            false, a[mi].v, false, b[ni].v, (short)0, acc[mi][ni], false, false);
  }

  // Epilogue.  C/D f32 layout: VGPR v, lane-half h -> M = v + 8h, N = lane&15.
#pragma unroll
  for (int mi = 0; mi < 4; ++mi) {
#pragma unroll
    for (int ni = 0; ni < 4; ++ni) {
      v8f av = acc[mi][ni];
#pragma unroll
      for (int vv = 0; vv < 8; ++vv) {
        const int rr = row0 + mi * 16 + vv + 8 * h;   // local row
        const int cc = col0 + ni * 16 + r;            // local col
        float val = av[vv];
        const size_t oidx = (size_t)z * bsO + (size_t)rr * N + cc;

        if constexpr (EPI == 0) {
          val += bias[cc];
          if (xres) val += xres[(size_t)rr * N + cc];
          outf[oidx] = val;
        } else if constexpr (EPI == 1) {
          val = fmaxf(val + bias[cc], 0.0f);
          unsigned short bv = f2bf(val);
          outb[oidx] = bv;                     // hb (N = 3072)
          if (cc < 1536) {                     // v^T[b][d][i], S = 2048
            const int bb = rr >> 11, ii = rr & 2047;
            vt[((size_t)bb * 1536 + cc) * 2048 + ii] = bv;
          }
        } else if constexpr (EPI == 2) {
          val = fmaxf(val + bias[cc], 0.0f);
          outb[oidx] = f2bf(val);
        } else if constexpr (EPI == 3) {
          outb[oidx] = f2bf(val);
        } else if constexpr (EPI == 4) {
          val = fmaxf(val * scale, 0.0f);
          val = val * val;
          outb[oidx] = f2bf(val);
        } else if constexpr (EPI == 5) {
          const int rg = z * 2048 + rr;        // global row
          const float g = bf2f(gateb[(size_t)rg * 3072 + 1536 + cc]);
          outb[oidx] = f2bf(val * g);
        }
      }
    }
  }
}

// ---------------------------------------------------------------------------
// LayerNorm over D=512, one block (256 threads) per row.
// ---------------------------------------------------------------------------
__global__ void __launch_bounds__(256)
ln_bf16_kernel(const float* __restrict__ y, const float* __restrict__ g,
               const float* __restrict__ b, unsigned short* __restrict__ outb) {
  __shared__ float ss[256], sq[256];
  const int row = blockIdx.x;
  const float* p = y + (size_t)row * 512;
  const float x0 = p[threadIdx.x];
  const float x1 = p[threadIdx.x + 256];
  ss[threadIdx.x] = x0 + x1;
  sq[threadIdx.x] = x0 * x0 + x1 * x1;
  __syncthreads();
  for (int st = 128; st > 0; st >>= 1) {
    if (threadIdx.x < st) {
      ss[threadIdx.x] += ss[threadIdx.x + st];
      sq[threadIdx.x] += sq[threadIdx.x + st];
    }
    __syncthreads();
  }
  const float mean = ss[0] * (1.0f / 512.0f);
  const float var  = sq[0] * (1.0f / 512.0f) - mean * mean;
  const float rstd = rsqrtf(var + 1e-5f);
  const int c0 = threadIdx.x, c1 = threadIdx.x + 256;
  outb[(size_t)row * 512 + c0] = f2bf((x0 - mean) * rstd * g[c0] + b[c0]);
  outb[(size_t)row * 512 + c1] = f2bf((x1 - mean) * rstd * g[c1] + b[c1]);
}

__global__ void __launch_bounds__(256)
ln_add_x_kernel(const float* __restrict__ t, const float* __restrict__ g,
                const float* __restrict__ b, const float* __restrict__ x,
                float* __restrict__ out) {
  __shared__ float ss[256], sq[256];
  const int row = blockIdx.x;
  const float* p = t + (size_t)row * 512;
  const float x0 = p[threadIdx.x];
  const float x1 = p[threadIdx.x + 256];
  ss[threadIdx.x] = x0 + x1;
  sq[threadIdx.x] = x0 * x0 + x1 * x1;
  __syncthreads();
  for (int st = 128; st > 0; st >>= 1) {
    if (threadIdx.x < st) {
      ss[threadIdx.x] += ss[threadIdx.x + st];
      sq[threadIdx.x] += sq[threadIdx.x + st];
    }
    __syncthreads();
  }
  const float mean = ss[0] * (1.0f / 512.0f);
  const float var  = sq[0] * (1.0f / 512.0f) - mean * mean;
  const float rstd = rsqrtf(var + 1e-5f);
  const int c0 = threadIdx.x, c1 = threadIdx.x + 256;
  const size_t base = (size_t)row * 512;
  out[base + c0] = (x0 - mean) * rstd * g[c0] + b[c0] + x[base + c0];
  out[base + c1] = (x1 - mean) * rstd * g[c1] + b[c1] + x[base + c1];
}

// ---------------------------------------------------------------------------
// Host-side orchestration
// ---------------------------------------------------------------------------
extern "C" void kernel_launch(void* const* d_in, const int* in_sizes, int n_in,
                              void* d_out, int out_size, void* d_ws, size_t ws_size,
                              hipStream_t stream) {
  constexpr int Bn  = 4;
  constexpr int S   = 2048;
  constexpr int D   = 512;
  constexpr int QKD = 256;
  constexpr int HID = 1536;
  constexpr int H2  = 3072;
  constexpr int BS  = Bn * S;   // 8192

  const float* x  = (const float*)d_in[0];
  const float* Wm = (const float*)d_in[1];
  const float* bm = (const float*)d_in[2];
  const float* g1 = (const float*)d_in[3];
  const float* b1 = (const float*)d_in[4];
  const float* Wh = (const float*)d_in[5];
  const float* bh = (const float*)d_in[6];
  const float* Wq = (const float*)d_in[7];
  const float* bq = (const float*)d_in[8];
  const float* Wb = (const float*)d_in[9];
  const float* Wo = (const float*)d_in[10];
  const float* bo = (const float*)d_in[11];
  const float* g2 = (const float*)d_in[12];
  const float* b2 = (const float*)d_in[13];
  float* out = (float*)d_out;

  char* w = (char*)d_ws;
  size_t off = 0;
  auto alloc = [&](size_t bytes) -> void* {
    void* p = w + off;
    off += (bytes + 255) & ~(size_t)255;
    return p;
  };

  unsigned short* xb      = (unsigned short*)alloc((size_t)BS * D * 2);
  unsigned short* wmb     = (unsigned short*)alloc((size_t)D * D * 2);
  unsigned short* whb     = (unsigned short*)alloc((size_t)H2 * D * 2);
  unsigned short* wqb     = (unsigned short*)alloc((size_t)QKD * D * 2);
  unsigned short* wbt     = (unsigned short*)alloc((size_t)QKD * QKD * 2);
  unsigned short* wob     = (unsigned short*)alloc((size_t)D * HID * 2);
  float*          y       = (float*)alloc((size_t)BS * D * 4);
  unsigned short* normedb = (unsigned short*)alloc((size_t)BS * D * 2);
  unsigned short* hb      = (unsigned short*)alloc((size_t)BS * H2 * 2);
  unsigned short* vt      = (unsigned short*)alloc((size_t)Bn * HID * S * 2);
  unsigned short* zb      = (unsigned short*)alloc((size_t)BS * QKD * 2);
  unsigned short* zwb     = (unsigned short*)alloc((size_t)BS * QKD * 2);
  unsigned short* ab      = (unsigned short*)alloc((size_t)Bn * S * S * 2);
  unsigned short* vgb     = (unsigned short*)alloc((size_t)BS * HID * 2);
  float*          t       = y;   // y fully consumed by LN1 before t is written

  // --- bf16 conversions ---
  cvt_bf16_kernel<<<1024, 256, 0, stream>>>(x,  xb,  (long long)BS * D);
  cvt_bf16_kernel<<<256,  256, 0, stream>>>(Wm, wmb, (long long)D * D);
  cvt_bf16_kernel<<<1024, 256, 0, stream>>>(Wh, whb, (long long)H2 * D);
  cvt_bf16_kernel<<<256,  256, 0, stream>>>(Wq, wqb, (long long)QKD * D);
  cvt_bf16_tr_kernel<<<(QKD * QKD + 255) / 256, 256, 0, stream>>>(Wb, wbt, QKD, QKD);
  cvt_bf16_kernel<<<1024, 256, 0, stream>>>(Wo, wob, (long long)D * HID);

  // --- 1. y = x + x @ Wm^T + bm ---
  gemm_wmma<0><<<dim3(D / 128, BS / 256, 1), 256, 0, stream>>>(
      xb, wmb, D, 0, 0, 0, bm, x, y, nullptr, nullptr, nullptr, 1.0f);

  // --- 2. normed = LN(y) -> bf16 ---
  ln_bf16_kernel<<<BS, 256, 0, stream>>>(y, g1, b1, normedb);

  // --- 3. h = relu(normed @ Wh^T + bh): hb (bf16) + v^T ---
  gemm_wmma<1><<<dim3(H2 / 128, BS / 256, 1), 256, 0, stream>>>(
      normedb, whb, D, 0, 0, 0, bh, nullptr, nullptr, hb, vt, nullptr, 1.0f);

  // --- 4. Z = relu(normed @ Wq^T + bq) ---
  gemm_wmma<2><<<dim3(QKD / 128, BS / 256, 1), 256, 0, stream>>>(
      normedb, wqb, D, 0, 0, 0, bq, nullptr, nullptr, zb, nullptr, nullptr, 1.0f);

  // --- 5. ZW = Z @ Wb  (Wb pre-transposed to N x K) ---
  gemm_wmma<3><<<dim3(QKD / 128, BS / 256, 1), 256, 0, stream>>>(
      zb, wbt, QKD, 0, 0, 0, nullptr, nullptr, nullptr, zwb, nullptr, nullptr, 1.0f);

  // --- 6. A = relu(ZW @ Z^T / S)^2  (batched over z) ---
  gemm_wmma<4><<<dim3(S / 128, S / 256, Bn), 256, 0, stream>>>(
      zwb, zb, QKD, (size_t)S * QKD, (size_t)S * QKD, (size_t)S * S,
      nullptr, nullptr, nullptr, ab, nullptr, nullptr, 1.0f / (float)S);

  // --- 7. Vg = (A @ v) * gate  (batched; B operand = v^T) ---
  gemm_wmma<5><<<dim3(HID / 128, S / 256, Bn), 256, 0, stream>>>(
      ab, vt, S, (size_t)S * S, (size_t)HID * S, (size_t)S * HID,
      nullptr, nullptr, nullptr, vgb, nullptr, hb, 1.0f);

  // --- 8. t = Vg @ Wo^T + bo ---
  gemm_wmma<0><<<dim3(D / 128, BS / 256, 1), 256, 0, stream>>>(
      vgb, wob, HID, 0, 0, 0, bo, nullptr, t, nullptr, nullptr, nullptr, 1.0f);

  // --- 9. out = LN(t) + x ---
  ln_add_x_kernel<<<BS, 256, 0, stream>>>(t, g2, b2, x, out);

  (void)in_sizes; (void)n_in; (void)out_size; (void)ws_size;
}
